// LSTM_VAE_15891378995816
// MI455X (gfx1250) — matrix operand order repaired and checked by reading, not compiled, not run on hardware
//
#include <hip/hip_runtime.h>
#include <hip/hip_bf16.h>
#include <math.h>

// ---- problem constants (match reference) ----
#define BB   128       // batch
#define TT   512       // seq len
#define II   64        // input size (== latent L)
#define HH   512       // hidden
#define LL   64        // latent
#define GG   2048      // 4*H
#define KC   576       // H + I  (combined GEMM K)
#define DEC  100       // decoder steps

typedef __attribute__((ext_vector_type(16))) _Float16 v16h;
typedef __attribute__((ext_vector_type(8)))  float    v8f;

union V16H { v16h v; unsigned int u[8]; };

__device__ __forceinline__ float fast_sigmoid(float x) {
    return 1.f / (1.f + __expf(-x));
}
__device__ __forceinline__ float fast_tanh(float x) {
    // tanh(x) = 1 - 2/(1 + e^{2x})
    return 1.f - 2.f / (1.f + __expf(2.f * x));
}

// =====================================================================
// Prep kernels (run once per launch; deterministic)
// =====================================================================

// dst[n*KC + k] = (k < H) ? Whh[n*H+k] : Wih[n*I + (k-H)]   (f32 -> f16)
__global__ void pack_comb_kernel(const float* __restrict__ Whh,
                                 const float* __restrict__ Wih,
                                 _Float16* __restrict__ dst) {
    int idx = blockIdx.x * blockDim.x + threadIdx.x;   // < GG*KC
    if (idx >= GG * KC) return;
    int n = idx / KC, k = idx % KC;
    float v = (k < HH) ? Whh[n * HH + k] : Wih[n * II + (k - HH)];
    dst[idx] = (_Float16)v;
}

__global__ void f32_to_f16_kernel(const float* __restrict__ src,
                                  _Float16* __restrict__ dst, int n) {
    int idx = blockIdx.x * blockDim.x + threadIdx.x;
    if (idx < n) dst[idx] = (_Float16)src[idx];
}

__global__ void add_bias_kernel(const float* __restrict__ a,
                                const float* __restrict__ b,
                                float* __restrict__ dst, int n) {
    int idx = blockIdx.x * blockDim.x + threadIdx.x;
    if (idx < n) dst[idx] = a[idx] + b[idx];
}

__global__ void zero_state_kernel(float* __restrict__ h, float* __restrict__ c,
                                  _Float16* __restrict__ h16) {
    int idx = blockIdx.x * blockDim.x + threadIdx.x;   // < BB*HH
    h[idx] = 0.f; c[idx] = 0.f; h16[idx] = (_Float16)0.f;
}

// =====================================================================
// WMMA gates GEMM:  C[128 x 2048] = [Ah | Ax](f16) x W^T + bias
//   W: 2048 x 576 f16 row-major.  8 waves/block, 1 tile/wave, 128 blocks.
//   Fully unrolled K loop (18 WMMAs) so the scheduler pipelines loads
//   ahead of the serially-dependent accumulator chain.
// =====================================================================
__global__ __launch_bounds__(256) void gemm_gates_kernel(
    const _Float16* __restrict__ Ah, int ldAh,     // columns [0, HH)
    const _Float16* __restrict__ Ax, int ldAx,     // columns [HH, KC)
    const _Float16* __restrict__ W,                // GG x KC
    const float* __restrict__ bias,                // GG (b_ih + b_hh)
    float* __restrict__ Cmat) {
    const int tid   = threadIdx.x;
    const int wave  = tid >> 5;
    const int lane  = tid & 31;
    const int tile  = blockIdx.x * 8 + wave;       // 0..1023
    const int m_tile = tile >> 7;                  // /128 n-tiles
    const int n_tile = tile & 127;

    const int half = lane >> 4;
    const int l15  = lane & 15;
    const int mA   = m_tile * 16 + l15;
    const int nB   = n_tile * 16 + l15;
    const int khB  = half * 16;

    v8f acc = {};
    const _Float16* arowH = Ah + mA * ldAh;
    const _Float16* arowX = Ax + mA * ldAx;
    const _Float16* wrow0 = W + nB * KC + khB;

#pragma unroll
    for (int kb = 0; kb < (KC >> 5); ++kb) {
        const int k0 = kb << 5;
        const _Float16* arow = (k0 < HH) ? (arowH + k0) : (arowX + (k0 - HH));
        // ---- A fragment: 16x32 f16, contiguous K (k = 16*half + s), pairs of K per VGPR ----
        V16H a;
#pragma unroll
        for (int v = 0; v < 8; ++v) {
            int kk = half * 16 + v * 2;
            a.u[v] = *(const unsigned int*)(arow + kk);
        }
        // ---- B fragment: 32x16 f16, K pairs per VGPR, N across lanes ----
        const _Float16* wrow = wrow0 + k0;
        V16H b;
#pragma unroll
        for (int v = 0; v < 8; ++v)
            b.u[v] = *(const unsigned int*)(wrow + 2 * v);
        if (kb + 1 < (KC >> 5))
            __builtin_prefetch(wrow + 32, 0, 1);   // global_prefetch_b8 next K-slice
        acc = __builtin_amdgcn_wmma_f32_16x16x32_f16(
            false, a.v, false, b.v, (short)0, acc, false, false);
    }
    // ---- epilogue: C/D layout -> row m = half*8 + r, col n = l15; fused bias ----
    const int n = n_tile * 16 + l15;
    const float bn = bias[n];
#pragma unroll
    for (int r = 0; r < 8; ++r) {
        int m = m_tile * 16 + half * 8 + r;
        Cmat[m * GG + n] = acc[r] + bn;
    }
}

// =====================================================================
// WMMA output projection: x_hat_t[128 x 64] = h16[128x512] x Wout^T + b
//   fused epilogue: f32 -> d_out slice, f16 -> next decoder input
//   32 tiles -> 4 blocks x 8 waves; fully unrolled (16 WMMAs)
// =====================================================================
__global__ __launch_bounds__(256) void gemm_out_kernel(
    const _Float16* __restrict__ A,      // 128 x 512
    const _Float16* __restrict__ W,      // 64 x 512
    const float* __restrict__ bias,      // 64
    float* __restrict__ outF,            // d_out base, elem (m,n) at m*DEC*II + t*II + n
    _Float16* __restrict__ xnext,        // 128 x 64
    int t) {
    const int tid  = threadIdx.x;
    const int wave = tid >> 5;
    const int lane = tid & 31;
    const int tile = blockIdx.x * 8 + wave;   // 0..31
    const int m_tile = tile >> 2;
    const int n_tile = tile & 3;

    const int half = lane >> 4;
    const int l15  = lane & 15;
    const int mA   = m_tile * 16 + l15;
    const int nB   = n_tile * 16 + l15;
    const int khB  = half * 16;

    v8f acc = {};
    const _Float16* arowB = A + mA * HH;
    const _Float16* wrow0 = W + nB * HH + khB;
#pragma unroll
    for (int kb = 0; kb < (HH >> 5); ++kb) {
        const int k0 = kb << 5;
        const _Float16* arow = arowB + k0;
        V16H a;
#pragma unroll
        for (int v = 0; v < 8; ++v) {
            int kk = half * 16 + v * 2;
            a.u[v] = *(const unsigned int*)(arow + kk);
        }
        const _Float16* wrow = wrow0 + k0;
        V16H b;
#pragma unroll
        for (int v = 0; v < 8; ++v)
            b.u[v] = *(const unsigned int*)(wrow + 2 * v);
        acc = __builtin_amdgcn_wmma_f32_16x16x32_f16(
            false, a.v, false, b.v, (short)0, acc, false, false);
    }
    const int n = n_tile * 16 + l15;
    const float bn = bias[n];
#pragma unroll
    for (int r = 0; r < 8; ++r) {
        int m = m_tile * 16 + half * 8 + r;
        float val = acc[r] + bn;
        outF[m * (DEC * II) + t * II + n] = val;
        xnext[m * II + n] = (_Float16)val;
    }
}

// =====================================================================
// LSTM pointwise cell (PyTorch gate order i,f,g,o); gates pre-biased
// =====================================================================
__global__ void lstm_cell_kernel(const float* __restrict__ gates,
                                 float* __restrict__ h, float* __restrict__ c,
                                 _Float16* __restrict__ h16) {
    int idx = blockIdx.x * blockDim.x + threadIdx.x;   // < BB*HH
    int b = idx >> 9, j = idx & (HH - 1);
    const float* g = gates + b * GG;
    float gi = g[j];
    float gf = g[HH + j];
    float gg = g[2 * HH + j];
    float go = g[3 * HH + j];
    float cs = fast_sigmoid(gf) * c[idx] + fast_sigmoid(gi) * fast_tanh(gg);
    float hs = fast_sigmoid(go) * fast_tanh(cs);
    c[idx] = cs;
    h[idx] = hs;
    h16[idx] = (_Float16)hs;
}

// =====================================================================
// Latent: mean/logvar (write to d_out tail) + z = mean + eps*exp(0.5*lv)
// =====================================================================
__global__ void latent_kernel(const float* __restrict__ h,
                              const float* __restrict__ Wm, const float* __restrict__ bm,
                              const float* __restrict__ Wv, const float* __restrict__ bv,
                              const float* __restrict__ eps,
                              float* __restrict__ outMean, float* __restrict__ outLogvar,
                              float* __restrict__ z) {
    int idx = blockIdx.x * blockDim.x + threadIdx.x;   // < BB*LL
    int b = idx >> 6, l = idx & 63;
    const float* hr = h + b * HH;
    const float* wm = Wm + l * HH;
    const float* wv = Wv + l * HH;
    float sm = 0.f, sv = 0.f;
    for (int k = 0; k < HH; ++k) {
        float hv = hr[k];
        sm += hv * wm[k];
        sv += hv * wv[k];
    }
    sm += bm[l]; sv += bv[l];
    outMean[idx] = sm;
    outLogvar[idx] = sv;
    z[idx] = sm + eps[idx] * __expf(0.5f * sv);
}

// h_dec = z @ W_init^T + b_init ; c_dec = 0 ; x0 = 0
__global__ void dec_init_kernel(const float* __restrict__ z,
                                const float* __restrict__ Wi, const float* __restrict__ bi,
                                float* __restrict__ hd, float* __restrict__ cd,
                                _Float16* __restrict__ hd16, _Float16* __restrict__ xd16) {
    int idx = blockIdx.x * blockDim.x + threadIdx.x;   // < BB*HH
    int b = idx >> 9, j = idx & (HH - 1);
    const float* zr = z + b * LL;
    const float* w  = Wi + j * LL;
    float s = bi[j];
    for (int l = 0; l < LL; ++l) s += zr[l] * w[l];
    hd[idx] = s;
    cd[idx] = 0.f;
    hd16[idx] = (_Float16)s;
    if (j < II) xd16[b * II + j] = (_Float16)0.f;
}

// =====================================================================
// Host-side orchestration
// =====================================================================
extern "C" void kernel_launch(void* const* d_in, const int* in_sizes, int n_in,
                              void* d_out, int out_size, void* d_ws, size_t ws_size,
                              hipStream_t stream) {
    (void)in_sizes; (void)n_in; (void)out_size; (void)ws_size;

    const float* sequences = (const float*)d_in[0];
    const float* eps       = (const float*)d_in[2];
    const float* W_ih_enc  = (const float*)d_in[3];
    const float* W_hh_enc  = (const float*)d_in[4];
    const float* b_ih_enc  = (const float*)d_in[5];
    const float* b_hh_enc  = (const float*)d_in[6];
    const float* W_mean    = (const float*)d_in[7];
    const float* b_mean    = (const float*)d_in[8];
    const float* W_logvar  = (const float*)d_in[9];
    const float* b_logvar  = (const float*)d_in[10];
    const float* W_init    = (const float*)d_in[11];
    const float* b_init    = (const float*)d_in[12];
    const float* W_ih_dec  = (const float*)d_in[13];
    const float* W_hh_dec  = (const float*)d_in[14];
    const float* b_ih_dec  = (const float*)d_in[15];
    const float* b_hh_dec  = (const float*)d_in[16];
    const float* W_out     = (const float*)d_in[17];
    const float* b_out     = (const float*)d_in[18];
    float* out = (float*)d_out;

    // ---- carve workspace (256B aligned chunks) ----
    size_t off = 0;
    char* base = (char*)d_ws;
    auto carve = [&](size_t bytes) -> void* {
        void* p = base + off;
        off += (bytes + 255) & ~(size_t)255;
        return p;
    };
    _Float16* xseq16   = (_Float16*)carve((size_t)BB * TT * II * 2);
    _Float16* WcombE16 = (_Float16*)carve((size_t)GG * KC * 2);
    _Float16* WcombD16 = (_Float16*)carve((size_t)GG * KC * 2);
    _Float16* Wout16   = (_Float16*)carve((size_t)II * HH * 2);
    float*    biasE    = (float*)carve((size_t)GG * 4);
    float*    biasD    = (float*)carve((size_t)GG * 4);
    float*    h        = (float*)carve((size_t)BB * HH * 4);
    float*    c        = (float*)carve((size_t)BB * HH * 4);
    _Float16* h16      = (_Float16*)carve((size_t)BB * HH * 2);
    float*    gates    = (float*)carve((size_t)BB * GG * 4);
    float*    z        = (float*)carve((size_t)BB * LL * 4);
    float*    hd       = (float*)carve((size_t)BB * HH * 4);
    float*    cd       = (float*)carve((size_t)BB * HH * 4);
    _Float16* hd16     = (_Float16*)carve((size_t)BB * HH * 2);
    _Float16* xd16     = (_Float16*)carve((size_t)BB * II * 2);

    // ---- one-time prep ----
    pack_comb_kernel<<<(GG * KC + 255) / 256, 256, 0, stream>>>(W_hh_enc, W_ih_enc, WcombE16);
    pack_comb_kernel<<<(GG * KC + 255) / 256, 256, 0, stream>>>(W_hh_dec, W_ih_dec, WcombD16);
    f32_to_f16_kernel<<<(II * HH + 255) / 256, 256, 0, stream>>>(W_out, Wout16, II * HH);
    f32_to_f16_kernel<<<(BB * TT * II + 255) / 256, 256, 0, stream>>>(sequences, xseq16, BB * TT * II);
    add_bias_kernel<<<(GG + 255) / 256, 256, 0, stream>>>(b_ih_enc, b_hh_enc, biasE, GG);
    add_bias_kernel<<<(GG + 255) / 256, 256, 0, stream>>>(b_ih_dec, b_hh_dec, biasD, GG);
    zero_state_kernel<<<(BB * HH) / 256, 256, 0, stream>>>(h, c, h16);

    // ---- encoder: 512 sequential steps ----
    for (int t = 0; t < TT; ++t) {
        gemm_gates_kernel<<<128, 256, 0, stream>>>(
            h16, HH, xseq16 + (size_t)t * II, TT * II, WcombE16, biasE, gates);
        lstm_cell_kernel<<<(BB * HH) / 256, 256, 0, stream>>>(gates, h, c, h16);
    }

    // ---- latent head: mean, logvar -> d_out tail; z -> ws ----
    float* outMean   = out + (size_t)BB * DEC * II;
    float* outLogvar = outMean + (size_t)BB * LL;
    latent_kernel<<<(BB * LL) / 256, 256, 0, stream>>>(
        h, W_mean, b_mean, W_logvar, b_logvar, eps, outMean, outLogvar, z);

    // ---- decoder init ----
    dec_init_kernel<<<(BB * HH) / 256, 256, 0, stream>>>(z, W_init, b_init, hd, cd, hd16, xd16);

    // ---- decoder: 100 autoregressive steps ----
    for (int t = 0; t < DEC; ++t) {
        gemm_gates_kernel<<<128, 256, 0, stream>>>(hd16, HH, xd16, II, WcombD16, biasD, gates);
        lstm_cell_kernel<<<(BB * HH) / 256, 256, 0, stream>>>(gates, hd, cd, hd16);
        gemm_out_kernel<<<4, 256, 0, stream>>>(hd16, Wout16, b_out, out, xd16, t);
    }
}
